// GCN3D_27616639713982
// MI455X (gfx1250) — matrix operand order repaired and checked by reading, not compiled, not run on hardware
//
// 3D-GCN (mesh classifier) forward for MI455X / gfx1250.
// f16 WMMA (v_wmma_f32_16x16x32_f16, f32 accumulate) for all per-layer feature
// GEMMs with 2x4 register tile blocking (8 WMMAs per 6KB of loads), fused VALU
// kernels for kNN, edge vectors, neighbor-gather/max epilogue, pooling, FC head.
#include <hip/hip_runtime.h>
#include <hip/hip_bf16.h>

#define BS 4
#define V1 2048
#define V2 1024
#define NB 20
#define MT 2   // 16-row tiles per wave  (macro-tile rows = 32)
#define NT 4   // 16-col tiles per wave  (macro-tile cols = 64)

typedef __attribute__((ext_vector_type(16))) _Float16 v16h;
typedef __attribute__((ext_vector_type(8)))  _Float16 v8h;
typedef __attribute__((ext_vector_type(8)))  float    v8f;

// ---------------------------------------------------------------------------
// kNN: one thread per (b, i); insertion top-NB of squared distances (self skipped;
// strict '<' keeps earliest index on ties, matching jax.lax.top_k stability).
// ---------------------------------------------------------------------------
__global__ void knn_kernel(const float* __restrict__ verts, int Vc,
                           int* __restrict__ nbr) {
  int t = blockIdx.x * blockDim.x + threadIdx.x;
  if (t >= BS * Vc) return;
  int b = t / Vc, i = t % Vc;
  const float* vb = verts + (size_t)b * Vc * 3;
  float xi = vb[i * 3 + 0], yi = vb[i * 3 + 1], zi = vb[i * 3 + 2];
  float bd[NB]; int bi[NB];
  for (int n = 0; n < NB; ++n) { bd[n] = 3.4e38f; bi[n] = 0; }
  for (int j = 0; j < Vc; ++j) {
    if (j == i) continue;
    float dx = vb[j * 3 + 0] - xi;
    float dy = vb[j * 3 + 1] - yi;
    float dz = vb[j * 3 + 2] - zi;
    float dist = dx * dx + dy * dy + dz * dz;
    if (dist < bd[NB - 1]) {
      int k = NB - 1;
      while (k > 0 && bd[k - 1] > dist) { bd[k] = bd[k - 1]; bi[k] = bi[k - 1]; --k; }
      bd[k] = dist; bi[k] = j;
    }
  }
  for (int n = 0; n < NB; ++n) nbr[(size_t)t * NB + n] = bi[n];
}

// Normalized edge vectors per (b, v, n).
__global__ void diffn_kernel(const float* __restrict__ verts,
                             const int* __restrict__ nbr, int Vc,
                             float* __restrict__ diffn) {
  long t = (long)blockIdx.x * blockDim.x + threadIdx.x;
  long total = (long)BS * Vc * NB;
  if (t >= total) return;
  int n = (int)(t % NB);
  long m = t / NB;            // b*Vc + v
  int v = (int)(m % Vc);
  long bBase = m - v;         // b*Vc
  int j = nbr[m * NB + n];
  const float* pj = verts + (bBase + j) * 3;
  const float* pv = verts + m * 3;
  float dx = pj[0] - pv[0], dy = pj[1] - pv[1], dz = pj[2] - pv[2];
  float nm = fmaxf(sqrtf(dx * dx + dy * dy + dz * dz), 1e-12f);
  float* o = diffn + (m * NB + n) * 3;
  o[0] = dx / nm; o[1] = dy / nm; o[2] = dz / nm;
}

// Column-normalize direction matrix d (3, out_c).
__global__ void norm_dirs_kernel(const float* __restrict__ d, int outc,
                                 float* __restrict__ dirn) {
  int c = blockIdx.x * blockDim.x + threadIdx.x;
  if (c >= outc) return;
  float a = d[c], b = d[outc + c], e = d[2 * outc + c];
  float n = fmaxf(sqrtf(a * a + b * b + e * e), 1e-12f);
  dirn[c] = a / n; dirn[outc + c] = b / n; dirn[2 * outc + c] = e / n;
}

// fmap (M, Cin) f32 -> A16 (M, Kp) f16, zero-padded K.
__global__ void convert_A_kernel(const float* __restrict__ fm, int Cin, int Kp,
                                 _Float16* __restrict__ A16, long total) {
  long t = (long)blockIdx.x * blockDim.x + threadIdx.x;
  if (t >= total) return;
  int k = (int)(t % Kp);
  long m = t / Kp;
  A16[t] = (k < Cin) ? (_Float16)fm[m * Cin + k] : (_Float16)0.0f;
}

// w (Cin, N) f32 -> Bt (N, Kp) f16 transposed, zero-padded K (contiguous K per column).
__global__ void convert_Bt_kernel(const float* __restrict__ w, int Cin, int N, int Kp,
                                  _Float16* __restrict__ Bt, long total) {
  long t = (long)blockIdx.x * blockDim.x + threadIdx.x;
  if (t >= total) return;
  int k = (int)(t % Kp);
  long n = t / Kp;
  Bt[t] = (k < Cin) ? (_Float16)w[(long)k * N + n] : (_Float16)0.0f;
}

// ---------------------------------------------------------------------------
// Register-blocked WMMA GEMM: C(M,N) = A(M,Kp) * Bt(N,Kp)^T + bias.
// One wave computes a (MT*16) x (NT*16) macro-tile: per K=32 step it loads
// MT A-fragments + NT B-fragments and issues MT*NT v_wmma_f32_16x16x32_f16,
// i.e. 0.75KB of loads per WMMA instead of 2KB for a 1x1 tile.
// Fragment layouts used here (C/D as in CDNA5 ISA §7.12.2):
//   A 16x32 f16: lane L holds row L%16; K = (L/16)*16 + {0..15} (this code's contiguous order, same as B).
//   B 32x16 f16: lane L holds col L%16; K = (L/16)*16 + {0..15} (this code's contiguous order, same as A).
//   C/D 16x16 f32: VGPR e, lane L -> M = (L/16)*8 + e, N = L%16.
// ---------------------------------------------------------------------------
__global__ void __launch_bounds__(256)
wmma_gemm_bias_kernel(const _Float16* __restrict__ A,
                      const _Float16* __restrict__ Bt,
                      const float* __restrict__ bias,
                      float* __restrict__ C, int M, int N, int Kp) {
  int wave = (int)((blockIdx.x * blockDim.x + threadIdx.x) >> 5);
  int lane = threadIdx.x & 31;
  int nbN = N >> 6;                       // macro-tiles along N (64 cols each)
  int nBlocks = (M >> 5) * nbN;           // macro-tiles along M (32 rows each)
  if (wave >= nBlocks) return;            // wave-uniform: EXEC stays all-1s for WMMA
  int bm = wave / nbN;
  int bn = wave - bm * nbN;
  int l16 = lane & 15;
  int hi  = lane >> 4;

  const _Float16* aPtr[MT];
  const _Float16* bPtr[NT];
#pragma unroll
  for (int i = 0; i < MT; ++i)
    aPtr[i] = A + (size_t)((bm * MT + i) * 16 + l16) * Kp + (hi << 4);
#pragma unroll
  for (int j = 0; j < NT; ++j)
    bPtr[j] = Bt + (size_t)((bn * NT + j) * 16 + l16) * Kp + (hi << 4);

  v8f acc[MT][NT] = {};
  for (int k0 = 0; k0 < Kp; k0 += 32) {
    v16h a[MT], b[NT];
#pragma unroll
    for (int i = 0; i < MT; ++i) {
      v8h lo  = *(const v8h*)(aPtr[i] + k0);        // K = k0 + hi*16 + 0..7
      v8h hi2 = *(const v8h*)(aPtr[i] + k0 + 8);    // K = k0 + hi*16 + 8..15
      a[i] = __builtin_shufflevector(lo, hi2, 0, 1, 2, 3, 4, 5, 6, 7,
                                              8, 9, 10, 11, 12, 13, 14, 15);
      __builtin_prefetch(aPtr[i] + k0 + 32, 0, 0);  // -> global_prefetch_b8
    }
#pragma unroll
    for (int j = 0; j < NT; ++j) {
      b[j] = *(const v16h*)(bPtr[j] + k0);          // K = k0 + hi*16 + 0..15
      __builtin_prefetch(bPtr[j] + k0 + 32, 0, 0);
    }
#pragma unroll
    for (int i = 0; i < MT; ++i)
#pragma unroll
      for (int j = 0; j < NT; ++j)
        acc[i][j] = __builtin_amdgcn_wmma_f32_16x16x32_f16(
            /*neg_a=*/false, a[i], /*neg_b=*/false, b[j],
            /*c_mod=*/(short)0, acc[i][j], /*reuse_a=*/false, /*reuse_b=*/false);
  }

#pragma unroll
  for (int i = 0; i < MT; ++i) {
    int mBase = (bm * MT + i) * 16 + (hi << 3);
#pragma unroll
    for (int j = 0; j < NT; ++j) {
      int col = (bn * NT + j) * 16 + l16;
      float bb = bias[col];
#pragma unroll
      for (int e = 0; e < 8; ++e)
        C[(size_t)(mBase + e) * N + col] = acc[i][j][e] + bb;
    }
  }
}

// Fused conv epilogue: out[b,v,c] = feat[b,v,c] + max_n( (diffn(b,v,n)·dirn[:,c]) *
// feat[b, nbr(b,v,n), out_c + c] ).
__global__ void act_fuse_kernel(const float* __restrict__ feat,
                                const int* __restrict__ nbr,
                                const float* __restrict__ diffn,
                                const float* __restrict__ dirn,
                                float* __restrict__ out, int Vc, int outc) {
  long t = (long)blockIdx.x * blockDim.x + threadIdx.x;
  long total = (long)BS * Vc * outc;
  if (t >= total) return;
  int c = (int)(t % outc);
  long m = t / outc;                 // b*Vc + v
  int v = (int)(m % Vc);
  long bBase = m - v;                // b*Vc
  float d0 = dirn[c], d1 = dirn[outc + c], d2 = dirn[2 * outc + c];
  const int* nb = nbr + m * NB;
  const float* df = diffn + m * NB * 3;
  int twoC = 2 * outc;
  float best = -3.4e38f;
#pragma unroll 4
  for (int n = 0; n < NB; ++n) {
    int j = nb[n];
    float th = df[n * 3 + 0] * d0 + df[n * 3 + 1] * d1 + df[n * 3 + 2] * d2;
    float sup = feat[(bBase + j) * (long)twoC + outc + c];
    best = fmaxf(best, th * sup);
  }
  out[m * (long)outc + c] = feat[m * (long)twoC + c] + best;
}

// Max over neighbors (full graph) evaluated only at strided vertices (fused subsample).
__global__ void pool_sub_kernel(const float* __restrict__ fm,
                                const int* __restrict__ nbr, int C,
                                float* __restrict__ out) {
  long t = (long)blockIdx.x * blockDim.x + threadIdx.x;
  long total = (long)BS * V2 * C;
  if (t >= total) return;
  int c = (int)(t % C);
  long m2 = t / C;
  int v2 = (int)(m2 % V2);
  int b = (int)(m2 / V2);
  const int* nb = nbr + ((long)b * V1 + 2 * v2) * NB;
  float best = -3.4e38f;
#pragma unroll 4
  for (int n = 0; n < NB; ++n)
    best = fmaxf(best, fm[((long)b * V1 + nb[n]) * C + c]);
  out[t] = best;
}

__global__ void subsample_verts_kernel(const float* __restrict__ verts,
                                       float* __restrict__ out) {
  int t = blockIdx.x * blockDim.x + threadIdx.x;
  if (t >= BS * V2 * 3) return;
  int k = t % 3;
  int r = t / 3;
  int v2 = r % V2, b = r / V2;
  out[t] = verts[((long)b * V1 + 2 * v2) * 3 + k];
}

// Global max over vertices.
__global__ void gmax_kernel(const float* __restrict__ fm, int C,
                            float* __restrict__ g) {
  int t = blockIdx.x * blockDim.x + threadIdx.x;
  if (t >= BS * C) return;
  int c = t % C, b = t / C;
  float best = -3.4e38f;
  for (int v = 0; v < V2; ++v)
    best = fmaxf(best, fm[((long)b * V2 + v) * C + c]);
  g[t] = best;
}

__global__ void fc1_kernel(const float* __restrict__ g, const float* __restrict__ w,
                           const float* __restrict__ bias, const float* __restrict__ bng,
                           const float* __restrict__ bnb, float* __restrict__ h) {
  int t = blockIdx.x * blockDim.x + threadIdx.x;
  if (t >= BS * 256) return;
  int j = t % 256, b = t / 256;
  const float* gr = g + b * 1024;
  float s = bias[j];
  for (int k = 0; k < 1024; ++k) s += gr[k] * w[k * 256 + j];
  s *= rsqrtf(1.0f + 1e-5f);
  s = s * bng[j] + bnb[j];
  h[t] = s > 0.0f ? s : 0.0f;
}

__global__ void fc2_kernel(const float* __restrict__ h, const float* __restrict__ w,
                           const float* __restrict__ bias, float* __restrict__ out) {
  int t = blockIdx.x * blockDim.x + threadIdx.x;
  if (t >= BS * 40) return;
  int j = t % 40, b = t / 40;
  float s = bias[j];
  for (int k = 0; k < 256; ++k) s += h[b * 256 + k] * w[k * 40 + j];
  out[t] = s;
}

// ---------------------------------------------------------------------------
// Host orchestration (graph-capture safe: only kernel launches on `stream`).
// ---------------------------------------------------------------------------
static inline int cdivl(long a, int b) { return (int)((a + b - 1) / b); }

static void run_conv(hipStream_t s, const float* fmIn, int Cin, int Vc,
                     const float* w, const float* bias, const float* d,
                     const int* nbr, const float* dif, int outc,
                     _Float16* A16, _Float16* Bt16, float* feat, float* dirn,
                     float* fmOut) {
  int M = BS * Vc, N = 2 * outc, Kp = (Cin + 31) & ~31;
  long ta = (long)M * Kp;
  convert_A_kernel<<<cdivl(ta, 256), 256, 0, s>>>(fmIn, Cin, Kp, A16, ta);
  long tb = (long)N * Kp;
  convert_Bt_kernel<<<cdivl(tb, 256), 256, 0, s>>>(w, Cin, N, Kp, Bt16, tb);
  norm_dirs_kernel<<<cdivl(outc, 64), 64, 0, s>>>(d, outc, dirn);
  int macroTiles = (M >> 5) * (N >> 6);   // (MT*16) x (NT*16) per wave
  wmma_gemm_bias_kernel<<<(macroTiles + 7) / 8, 256, 0, s>>>(A16, Bt16, bias, feat,
                                                             M, N, Kp);
  long tc = (long)M * outc;
  act_fuse_kernel<<<cdivl(tc, 256), 256, 0, s>>>(feat, nbr, dif, dirn, fmOut, Vc, outc);
}

extern "C" void kernel_launch(void* const* d_in, const int* in_sizes, int n_in,
                              void* d_out, int out_size, void* d_ws, size_t ws_size,
                              hipStream_t stream) {
  (void)in_sizes; (void)n_in; (void)out_size; (void)ws_size;
  const float* vertices = (const float*)d_in[0];
  const float *W[12], *Bp[12], *Dp[12];
  for (int i = 0; i < 12; ++i) {
    W[i]  = (const float*)d_in[1 + 3 * i];
    Bp[i] = (const float*)d_in[2 + 3 * i];
    Dp[i] = (const float*)d_in[3 + 3 * i];
  }
  static const int IC[12] = {3, 32, 64, 64, 64, 64, 64, 128, 256, 1024, 1024, 1024};
  static const int OC[12] = {32, 64, 64, 64, 64, 64, 128, 256, 1024, 1024, 1024, 1024};
  const float* fc1_w = (const float*)d_in[37];
  const float* fc1_b = (const float*)d_in[38];
  const float* bn_g  = (const float*)d_in[39];
  const float* bn_b  = (const float*)d_in[40];
  const float* fc2_w = (const float*)d_in[41];
  const float* fc2_b = (const float*)d_in[42];

  // Bump-allocate workspace (all offsets 256B-aligned; ~85 MB total).
  char* p = (char*)d_ws;
  auto alloc = [&](size_t bytes) -> void* {
    void* r = (void*)p;
    p += (bytes + 255) & ~(size_t)255;
    return r;
  };
  int*      nbr1   = (int*)     alloc((size_t)BS * V1 * NB * 4);
  int*      nbr2   = (int*)     alloc((size_t)BS * V2 * NB * 4);
  float*    dif1   = (float*)   alloc((size_t)BS * V1 * NB * 3 * 4);
  float*    dif2   = (float*)   alloc((size_t)BS * V2 * NB * 3 * 4);
  float*    dirn   = (float*)   alloc((size_t)3 * 1024 * 4);
  float*    fmA    = (float*)   alloc((size_t)BS * V2 * 1024 * 4);
  float*    fmB    = (float*)   alloc((size_t)BS * V2 * 1024 * 4);
  float*    feat   = (float*)   alloc((size_t)BS * V2 * 2048 * 4);
  _Float16* A16    = (_Float16*)alloc((size_t)BS * V2 * 1024 * 2);
  _Float16* Bt16   = (_Float16*)alloc((size_t)2048 * 1024 * 2);
  float*    verts2 = (float*)   alloc((size_t)BS * V2 * 3 * 4);
  float*    g      = (float*)   alloc((size_t)BS * 1024 * 4);
  float*    h      = (float*)   alloc((size_t)BS * 256 * 4);

  // Stage 1: full-resolution graph (V=2048).
  knn_kernel<<<cdivl((long)BS * V1, 128), 128, 0, stream>>>(vertices, V1, nbr1);
  diffn_kernel<<<cdivl((long)BS * V1 * NB, 256), 256, 0, stream>>>(vertices, nbr1, V1, dif1);

  const float* fmIn = vertices;
  float* bufs[2] = {fmA, fmB};
  int cur = 0;
  for (int L = 0; L < 6; ++L) {
    run_conv(stream, fmIn, IC[L], V1, W[L], Bp[L], Dp[L], nbr1, dif1, OC[L],
             A16, Bt16, feat, dirn, bufs[cur]);
    fmIn = bufs[cur];
    cur ^= 1;
  }

  // Neighbor-max pool fused with stride-2 subsample; rebuild graph at V=1024.
  pool_sub_kernel<<<cdivl((long)BS * V2 * 64, 256), 256, 0, stream>>>(fmIn, nbr1, 64, bufs[cur]);
  fmIn = bufs[cur];
  cur ^= 1;
  subsample_verts_kernel<<<cdivl((long)BS * V2 * 3, 256), 256, 0, stream>>>(vertices, verts2);
  knn_kernel<<<cdivl((long)BS * V2, 128), 128, 0, stream>>>(verts2, V2, nbr2);
  diffn_kernel<<<cdivl((long)BS * V2 * NB, 256), 256, 0, stream>>>(verts2, nbr2, V2, dif2);

  // Stage 2: V=1024, up to 1024 channels (WMMA-dominated).
  for (int L = 6; L < 12; ++L) {
    run_conv(stream, fmIn, IC[L], V2, W[L], Bp[L], Dp[L], nbr2, dif2, OC[L],
             A16, Bt16, feat, dirn, bufs[cur]);
    fmIn = bufs[cur];
    cur ^= 1;
  }

  // Head: global max + FC(1024->256)+BN+ReLU + FC(256->40).
  gmax_kernel<<<cdivl((long)BS * 1024, 256), 256, 0, stream>>>(fmIn, 1024, g);
  fc1_kernel<<<cdivl((long)BS * 256, 256), 256, 0, stream>>>(g, fc1_w, fc1_b, bn_g, bn_b, h);
  fc2_kernel<<<cdivl((long)BS * 40, 64), 64, 0, stream>>>(h, fc2_w, fc2_b, (float*)d_out);
}
